// SpatialAttentionLayer_15238543966276
// MI455X (gfx1250) — matrix operand order repaired and checked by reading, not compiled, not run on hardware
//
#include <hip/hip_runtime.h>
#include <hip/hip_bf16.h>

typedef __attribute__((ext_vector_type(16))) __bf16 v16bf;
typedef __attribute__((ext_vector_type(8)))  float  v8f;

#define T_DIM   120
#define B_DIM   64
#define N_JOINT 20
#define D_MODEL 256
#define H_HEADS 8
#define F_HEAD  32
#define BT      16                      // batch tile per block
#define KB      8                       // 256 / 32 k-steps

#define WQ_FRAGS (H_HEADS * N_JOINT * 2 * KB)   // 2560 fragments
#define WK_FRAGS (H_HEADS * 2 * KB)             // 128 fragments
#define FRAG_HALF 512                           // 32 lanes * 16 bf16 per fragment

#define X_STRIDE   528                  // bytes per x row in LDS (256 bf16 + 16B pad)
#define QKV_STRIDE 33                   // floats per qkv row in LDS (32 + 1 pad)
#define X_BYTES    (320 * X_STRIDE)               // 168960
#define QKV_BYTES  (320 * QKV_STRIDE * 4)         // 42240
#define LDS_TOTAL  (X_BYTES + 3 * QKV_BYTES)      // 295680

__device__ __forceinline__ unsigned short f2bf(float x) {
    unsigned u = __float_as_uint(x);
    u += 0x7FFFu + ((u >> 16) & 1u);            // round-to-nearest-even
    return (unsigned short)(u >> 16);
}

// ---------------------------------------------------------------------------
// Prep: convert f32 weights to bf16 WMMA B-fragments in workspace.
// Fragment = 32x16 (KxN) tile; lane l<16 holds col f=l, K=0..15;
// lane l>=16 holds col f=l-16, K=16..31 (16 contiguous halves per lane).
// Fragment order: Wq [h][n][ftile][kb], then Wk [h][ftile][kb], then Wv.
// ---------------------------------------------------------------------------
__global__ void prep_weights_kernel(const float* __restrict__ Wq,
                                    const float* __restrict__ Wk,
                                    const float* __restrict__ Wv,
                                    __bf16* __restrict__ ws)
{
    int gid  = blockIdx.x * blockDim.x + threadIdx.x;
    int frag = gid >> 5;
    int lane = gid & 31;
    if (frag >= WQ_FRAGS + 2 * WK_FRAGS) return;

    int f_local = lane & 15;
    const float* src;
    if (frag < WQ_FRAGS) {
        int kb = frag & 7, ft = (frag >> 3) & 1, hn = frag >> 4;   // hn = h*20+n
        int kbase = kb * 32 + (lane >> 4) * 16;
        src = Wq + ((size_t)(hn * F_HEAD + ft * 16 + f_local)) * D_MODEL + kbase;
    } else if (frag < WQ_FRAGS + WK_FRAGS) {
        int fr = frag - WQ_FRAGS;
        int kb = fr & 7, ft = (fr >> 3) & 1, h = fr >> 4;
        int kbase = kb * 32 + (lane >> 4) * 16;
        src = Wk + ((size_t)(h * F_HEAD + ft * 16 + f_local)) * D_MODEL + kbase;
    } else {
        int fr = frag - WQ_FRAGS - WK_FRAGS;
        int kb = fr & 7, ft = (fr >> 3) & 1, h = fr >> 4;
        int kbase = kb * 32 + (lane >> 4) * 16;
        src = Wv + ((size_t)(h * F_HEAD + ft * 16 + f_local)) * D_MODEL + kbase;
    }

    unsigned pk[8];
    #pragma unroll
    for (int i = 0; i < 8; ++i)
        pk[i] = (unsigned)f2bf(src[2 * i]) | ((unsigned)f2bf(src[2 * i + 1]) << 16);

    unsigned* dst = (unsigned*)(ws + (size_t)frag * FRAG_HALF + lane * 16);
    *(uint4*)(dst)     = make_uint4(pk[0], pk[1], pk[2], pk[3]);
    *(uint4*)(dst + 4) = make_uint4(pk[4], pk[5], pk[6], pk[7]);
}

// ---------------------------------------------------------------------------
// Main kernel: one block per (t, batch-tile of 16). x staged once in LDS as
// bf16, reused for 8 heads x 3 projections via WMMA (q/k/v fused per tile so
// each A fragment is loaded from LDS once per 3 WMMAs); softmax in VALU.
// ---------------------------------------------------------------------------
__global__ __launch_bounds__(256, 1)
void spatial_attn_kernel(const float* __restrict__ x,
                         const float* __restrict__ bk,
                         const float* __restrict__ bv,
                         const float* __restrict__ bq,
                         const __bf16* __restrict__ wfrag,
                         float* __restrict__ out)
{
    extern __shared__ char smem[];
    char*  Xs = smem;                                  // [row=n*16+b][256 bf16], stride 528B
    float* qS = (float*)(smem + X_BYTES);
    float* kS = (float*)(smem + X_BYTES + QKV_BYTES);
    float* vS = (float*)(smem + X_BYTES + 2 * QKV_BYTES);

    const int t     = blockIdx.y;
    const int btile = blockIdx.x;
    const int tid   = threadIdx.x;
    const int lane  = tid & 31;
    const int wave  = tid >> 5;

    // ---- Stage x tile: f32 global -> bf16 LDS, layout [n][b][d] ----
    for (int it = tid; it < 320 * 64; it += 256) {     // 64 float4 per 256-wide row
        int row = it >> 6;                             // row = n*16 + b
        int d4  = it & 63;
        int n = row >> 4, b = row & 15;
        const float4 v = *(const float4*)(
            x + (size_t)(t * B_DIM + btile * BT + b) * (N_JOINT * D_MODEL)
              + n * D_MODEL + d4 * 4);
        unsigned lo = (unsigned)f2bf(v.x) | ((unsigned)f2bf(v.y) << 16);
        unsigned hi = (unsigned)f2bf(v.z) | ((unsigned)f2bf(v.w) << 16);
        *(uint2*)(Xs + (size_t)row * X_STRIDE + d4 * 8) = make_uint2(lo, hi);
    }
    __syncthreads();

    const float scale = 0.17677669529663687f;          // 1/sqrt(32)

    for (int h = 0; h < H_HEADS; ++h) {
        // ---- Projections: 40 tiles (20 joints x 2 f-tiles), q/k/v fused ----
        // Tile rows = 16 batches of joint n -> one A fragment feeds 3 WMMAs.
        for (int tile = wave; tile < 40; tile += 8) {  // 5 tiles/wave, uniform
            int n  = tile >> 1;
            int ft = tile & 1;

            const __bf16* qfr = wfrag + (size_t)(((h * 20 + n) * 2 + ft) * 8) * FRAG_HALF;
            const __bf16* kfr = wfrag + (size_t)(WQ_FRAGS + (h * 2 + ft) * 8) * FRAG_HALF;
            const __bf16* vfr = wfrag + (size_t)(WQ_FRAGS + WK_FRAGS + (h * 2 + ft) * 8) * FRAG_HALF;

            // A-fragment: lane m -> row m, K{0-15};
            // lane m+16 -> row m, K{16-31}
            const char* arow = Xs + (size_t)(n * 16 + (lane & 15)) * X_STRIDE
                                  + ((lane >> 4) ? 32 : 0);

            v8f accq = {}, acck = {}, accv = {};
            #pragma unroll
            for (int kb = 0; kb < 8; ++kb) {
                union { uint4 u[2]; v16bf v; } A, Bq, Bk, Bv;
                A.u[0] = *(const uint4*)(arow + kb * 64);
                A.u[1] = *(const uint4*)(arow + kb * 64 + 16);
                const char* qp = (const char*)(qfr + (size_t)kb * FRAG_HALF) + lane * 32;
                const char* kp = (const char*)(kfr + (size_t)kb * FRAG_HALF) + lane * 32;
                const char* vp = (const char*)(vfr + (size_t)kb * FRAG_HALF) + lane * 32;
                Bq.u[0] = *(const uint4*)qp;  Bq.u[1] = *(const uint4*)(qp + 16);
                Bk.u[0] = *(const uint4*)kp;  Bk.u[1] = *(const uint4*)(kp + 16);
                Bv.u[0] = *(const uint4*)vp;  Bv.u[1] = *(const uint4*)(vp + 16);
                accq = __builtin_amdgcn_wmma_f32_16x16x32_bf16(
                         false, A.v, false, Bq.v, (short)0, accq, false, false);
                acck = __builtin_amdgcn_wmma_f32_16x16x32_bf16(
                         false, A.v, false, Bk.v, (short)0, acck, false, false);
                accv = __builtin_amdgcn_wmma_f32_16x16x32_bf16(
                         false, A.v, false, Bv.v, (short)0, accv, false, false);
            }

            // Bias + scatter D fragments (lane l: col f, rows rr / rr+8)
            int f     = ft * 16 + (lane & 15);
            int brow0 = (lane < 16) ? 0 : 8;
            float biasq = bq[(h * 20 + n) * 32 + f];
            float biask = bk[h * 32 + f];
            float biasv = bv[h * 32 + f];
            #pragma unroll
            for (int rr = 0; rr < 8; ++rr) {
                size_t rbase = (size_t)(n * 16 + brow0 + rr) * QKV_STRIDE + f;
                qS[rbase] = accq[rr] + biasq;
                kS[rbase] = acck[rr] + biask;
                vS[rbase] = accv[rr] + biasv;
            }
        }
        __syncthreads();

        // ---- Attention per (n,b) row: N=20 softmax + AV in f32 VALU ----
        for (int rix = tid; rix < 320; rix += 256) {
            int n = rix >> 4, b = rix & 15;

            float qr[32];
            #pragma unroll
            for (int f = 0; f < 32; ++f)
                qr[f] = qS[(size_t)rix * QKV_STRIDE + f];

            float lg[20], mx = -1e30f;
            #pragma unroll
            for (int m = 0; m < 20; ++m) {
                const float* kr = kS + (size_t)(m * 16 + b) * QKV_STRIDE;
                float d = 0.f;
                #pragma unroll
                for (int f = 0; f < 32; ++f) d += qr[f] * kr[f];
                d *= scale;
                lg[m] = d;
                mx = fmaxf(mx, d);
            }
            float sum = 0.f;
            #pragma unroll
            for (int m = 0; m < 20; ++m) { float e = __expf(lg[m] - mx); lg[m] = e; sum += e; }
            float inv = 1.f / sum;

            float o[32];
            #pragma unroll
            for (int f = 0; f < 32; ++f) o[f] = 0.f;
            #pragma unroll
            for (int m = 0; m < 20; ++m) {
                const float* vr = vS + (size_t)(m * 16 + b) * QKV_STRIDE;
                float pw = lg[m] * inv;
                #pragma unroll
                for (int f = 0; f < 32; ++f) o[f] += pw * vr[f];
            }

            float* op = out + (size_t)((t * B_DIM + btile * BT + b) * N_JOINT + n) * D_MODEL
                            + h * F_HEAD;
            #pragma unroll
            for (int f = 0; f < 32; f += 4)
                *(float4*)(op + f) = make_float4(o[f], o[f + 1], o[f + 2], o[f + 3]);
        }
        __syncthreads();                               // qkv LDS reused next head
    }
}

// ---------------------------------------------------------------------------
extern "C" void kernel_launch(void* const* d_in, const int* in_sizes, int n_in,
                              void* d_out, int out_size, void* d_ws, size_t ws_size,
                              hipStream_t stream)
{
    (void)in_sizes; (void)n_in; (void)out_size; (void)ws_size;

    const float* x  = (const float*)d_in[0];
    const float* Wk = (const float*)d_in[1];
    const float* bk = (const float*)d_in[2];
    const float* Wv = (const float*)d_in[3];
    const float* bv = (const float*)d_in[4];
    const float* Wq = (const float*)d_in[5];
    const float* bq = (const float*)d_in[6];
    float* out      = (float*)d_out;
    __bf16* wfrag   = (__bf16*)d_ws;                  // ~2.9 MB of bf16 fragments

    int frag_threads = (WQ_FRAGS + 2 * WK_FRAGS) * 32;    // 90112
    prep_weights_kernel<<<(frag_threads + 255) / 256, 256, 0, stream>>>(Wq, Wk, Wv, wfrag);

    dim3 grid(B_DIM / BT, T_DIM);                     // 4 x 120 = 480 blocks
    spatial_attn_kernel<<<grid, 256, LDS_TOTAL, stream>>>(x, bk, bv, bq, wfrag, out);
}
